// QuantizationLayer_3264175145090
// MI455X (gfx1250) — matrix operand order repaired and checked by reading, not compiled, not run on hardware
//
#include <hip/hip_runtime.h>

// Residual multi-level VQ for MI455X (gfx1250, wave32, WMMA).
// B=32768 rows, L=4 levels, K=2048 codes, D=512 dims.
// Distance GEMM as split-bf16 (hi+lo): 3x v_wmma_f32_16x16x32_bf16 per
// 32-wide K chunk => ~f32 accuracy at bf16 matrix-core rate.
// Codebook tiles are staged block-wide into LDS with gfx1250 async LDS-DMA
// (global_load_async_to_lds_b128 + s_wait_asynccnt), double-buffered.

#define B_ 32768
#define L_ 4
#define K_ 2048
#define D_ 512
#define NTPG 4            // N-tiles (16 codes each) per group => 64 codes
#define MT 2              // M-tiles (16 rows each) per wave => 32 rows
#define SLICE_K 64        // K-elements staged per LDS slice
#define NSLICE (D_ / SLICE_K)
#define ROWS_E 72         // padded LDS row stride in ushort (144B, bank-safe)

typedef __attribute__((ext_vector_type(16))) __bf16 v16bf;
typedef __attribute__((ext_vector_type(8)))  __bf16 v8bf;
typedef __attribute__((ext_vector_type(8)))  float  v8f;

__device__ __forceinline__ unsigned short f32_to_bf16u(float f) {
    unsigned u = __float_as_uint(f);
    u += 0x7FFFu + ((u >> 16) & 1u);     // round-to-nearest-even
    return (unsigned short)(u >> 16);
}
__device__ __forceinline__ float bf16u_to_f32(unsigned short h) {
    return __uint_as_float(((unsigned)h) << 16);
}

// gfx1250 async LDS DMA: copies 16B from global to LDS, tracked by ASYNCcnt.
__device__ __forceinline__ void async_load_b128(unsigned lds_off,
                                                const void* gptr) {
    asm volatile("global_load_async_to_lds_b128 %0, %1, off"
                 :: "v"(lds_off),
                    "v"((unsigned long long)(size_t)gptr)
                 : "memory");
}

// ---------------- pre-pass kernels ----------------

__global__ __launch_bounds__(256) void prep_cb_kernel(
    const float* __restrict__ cb, unsigned short* __restrict__ hi,
    unsigned short* __restrict__ lo) {
    long i = (long)blockIdx.x * blockDim.x + threadIdx.x;
    if (i >= (long)L_ * K_ * D_) return;
    float f = cb[i];
    unsigned short h = f32_to_bf16u(f);
    hi[i] = h;
    lo[i] = f32_to_bf16u(f - bf16u_to_f32(h));
}

__global__ __launch_bounds__(256) void prep_norm_kernel(
    const float* __restrict__ cb, float* __restrict__ cbnorm,
    int* __restrict__ counts, float* __restrict__ out_count) {
    int i = blockIdx.x * blockDim.x + threadIdx.x;
    if (i == 0) *out_count = 0.0f;
    if (i >= L_ * K_) return;
    const float* row = cb + (long)i * D_;
    float s = 0.0f;
    for (int d = 0; d < D_; ++d) s = __builtin_fmaf(row[d], row[d], s);
    cbnorm[i] = s;
    counts[i] = 0;
}

__global__ __launch_bounds__(256) void prep_x_kernel(
    const float* __restrict__ x, float* __restrict__ resid,
    unsigned short* __restrict__ rhi, unsigned short* __restrict__ rlo,
    float* __restrict__ zacc) {
    long i = (long)blockIdx.x * blockDim.x + threadIdx.x;
    if (i >= (long)B_ * D_) return;
    float f = x[i];
    resid[i] = f;
    zacc[i]  = 0.0f;
    unsigned short h = f32_to_bf16u(f);
    rhi[i] = h;
    rlo[i] = f32_to_bf16u(f - bf16u_to_f32(h));
}

// ---------------- per-level kernels ----------------

// WMMA distance + argmin. Wave owns 32 rows (2 M-tiles); block = 8 waves =
// 256 rows. Per 64-code group, the block double-buffers 64-wide K slices of
// the (hi,lo) codebook into LDS with async LDS-DMA; every wave consumes the
// shared tile via ds_load_b128.
// d_eff[m][n] = ||cb_n||^2 - 2 * <x_m, cb_n>   (row norm is argmin-invariant)
__global__ __launch_bounds__(256) void vq_argmin_kernel(
    const __bf16* __restrict__ xhi, const __bf16* __restrict__ xlo,
    const unsigned short* __restrict__ cbhi,
    const unsigned short* __restrict__ cblo,
    const float* __restrict__ cbnorm,
    int* __restrict__ idx_out, int* __restrict__ counts,
    float* __restrict__ codes_out /* pre-offset by level */) {
    // rows 0..63: hi of codes 0..63 ; rows 64..127: lo of codes 0..63
    __shared__ __align__(32) unsigned short smem[2][128 * ROWS_E];

    const int tid  = threadIdx.x;
    const int lane = tid & 31;
    const int wave = tid >> 5;
    const int m0   = (blockIdx.x * 8 + wave) * (16 * MT);
    const int nloc = lane & 15;   // N within tile (B/C layout), also A row
    const int half = lane >> 4;   // K sub-group / M offset selector

    long arow[MT];
#pragma unroll
    for (int mt = 0; mt < MT; ++mt)
        arow[mt] = (long)(m0 + mt * 16 + nloc) * D_;

    const unsigned sbase = (unsigned)(unsigned long long)(void*)&smem[0][0];
    const unsigned bufbytes = 128 * ROWS_E * 2;

    float best[MT][8];
    int   bestn[MT][8];
#pragma unroll
    for (int mt = 0; mt < MT; ++mt)
#pragma unroll
        for (int v = 0; v < 8; ++v) { best[mt][v] = __builtin_inff(); bestn[mt][v] = 0; }

    union AV { v16bf v; v8bf h[2]; };

    for (int g = 0; g < K_ / (16 * NTPG); ++g) {
        const int nbase = g * 16 * NTPG;
        v8f acc[MT][NTPG];
#pragma unroll
        for (int mt = 0; mt < MT; ++mt)
#pragma unroll
            for (int t = 0; t < NTPG; ++t)
#pragma unroll
                for (int e = 0; e < 8; ++e) acc[mt][t][e] = 0.0f;

        // prime slice 0 into buffer 0 (4 async b128 per thread)
        {
#pragma unroll
            for (int j = 0; j < 4; ++j) {
                int c = tid + j * 256;              // 1024 16B chunks
                int row = c >> 3, off = c & 7;
                unsigned lds_off = sbase + row * (ROWS_E * 2) + off * 16;
                const unsigned short* src = (row < 64) ? cbhi : cblo;
                long elem = (long)(nbase + (row & 63)) * D_ + off * 8;
                async_load_b128(lds_off, src + elem);
            }
        }

#pragma unroll
        for (int s = 0; s < NSLICE; ++s) {
            const int cur = s & 1;
            // prefetch next slice into the other buffer
            if (s < NSLICE - 1) {
#pragma unroll
                for (int j = 0; j < 4; ++j) {
                    int c = tid + j * 256;
                    int row = c >> 3, off = c & 7;
                    unsigned lds_off = sbase + (cur ^ 1) * bufbytes +
                                       row * (ROWS_E * 2) + off * 16;
                    const unsigned short* src = (row < 64) ? cbhi : cblo;
                    long elem = (long)(nbase + (row & 63)) * D_ +
                                (s + 1) * SLICE_K + off * 8;
                    async_load_b128(lds_off, src + elem);
                }
                asm volatile("s_wait_asynccnt 0x4" ::: "memory");
            } else {
                asm volatile("s_wait_asynccnt 0x0" ::: "memory");
            }
            __syncthreads();   // slice s visible to all waves

            const unsigned short* scur = &smem[cur][0];
#pragma unroll
            for (int kkl = 0; kkl < SLICE_K; kkl += 32) {
                const int kglob = s * SLICE_K + kkl;
                // A tiles 16x32 bf16 (lanes 0-15: K 0..7|16..23,
                //                     lanes 16-31: K 8..15|24..31)
                AV ahi[MT], alo[MT];
#pragma unroll
                for (int mt = 0; mt < MT; ++mt) {
                    const __bf16* pa = xhi + arow[mt] + kglob + 8 * half;
                    ahi[mt].h[0] = *(const v8bf*)(pa);
                    ahi[mt].h[1] = *(const v8bf*)(pa + 16);
                    const __bf16* pl = xlo + arow[mt] + kglob + 8 * half;
                    alo[mt].h[0] = *(const v8bf*)(pl);
                    alo[mt].h[1] = *(const v8bf*)(pl + 16);
                }
#pragma unroll
                for (int t = 0; t < NTPG; ++t) {
                    // B tile 32x16: lane n holds K 8*half..+7 | 16+8*half..+7 of its code
                    const unsigned short* bp =
                        scur + (t * 16 + nloc) * ROWS_E + kkl + 8 * half;
                    AV bh, bl;
                    bh.h[0] = *(const v8bf*)(bp);
                    bh.h[1] = *(const v8bf*)(bp + 16);
                    bl.h[0] = *(const v8bf*)(bp + 64 * ROWS_E);
                    bl.h[1] = *(const v8bf*)(bp + 64 * ROWS_E + 16);
#pragma unroll
                    for (int mt = 0; mt < MT; ++mt) {
                        acc[mt][t] = __builtin_amdgcn_wmma_f32_16x16x32_bf16(
                            false, ahi[mt].v, false, bh.v, (short)0,
                            acc[mt][t], false, false);
                        acc[mt][t] = __builtin_amdgcn_wmma_f32_16x16x32_bf16(
                            false, ahi[mt].v, false, bl.v, (short)0,
                            acc[mt][t], false, false);
                        acc[mt][t] = __builtin_amdgcn_wmma_f32_16x16x32_bf16(
                            false, alo[mt].v, false, bh.v, (short)0,
                            acc[mt][t], false, false);
                    }
                }
            }
            __syncthreads();   // all waves done reading before next overwrite
        }

        // fold group into running per-lane argmin (lane = fixed n,
        // 8 m-values per M-tile: m = v + 8*half)
#pragma unroll
        for (int t = 0; t < NTPG; ++t) {
            const int n = nbase + t * 16 + nloc;
            const float cn = cbnorm[n];
#pragma unroll
            for (int mt = 0; mt < MT; ++mt)
#pragma unroll
                for (int v = 0; v < 8; ++v) {
                    float dd = __builtin_fmaf(-2.0f, acc[mt][t][v], cn);
                    if (dd < best[mt][v]) { best[mt][v] = dd; bestn[mt][v] = n; }
                }
        }
    }

    // reduce min over the 16 lanes sharing each M row (xor stays in half)
#pragma unroll
    for (int off = 1; off < 16; off <<= 1) {
#pragma unroll
        for (int mt = 0; mt < MT; ++mt)
#pragma unroll
            for (int v = 0; v < 8; ++v) {
                float od = __shfl_xor(best[mt][v], off, 32);
                int   oi = __shfl_xor(bestn[mt][v], off, 32);
                if (od < best[mt][v] ||
                    (od == best[mt][v] && oi < bestn[mt][v])) {
                    best[mt][v] = od; bestn[mt][v] = oi;
                }
            }
    }

    if (nloc == 0) {
#pragma unroll
        for (int mt = 0; mt < MT; ++mt)
#pragma unroll
            for (int v = 0; v < 8; ++v) {
                int row = m0 + mt * 16 + half * 8 + v;
                idx_out[row] = bestn[mt][v];
                codes_out[(long)row * L_] = (float)bestn[mt][v];
                atomicAdd(&counts[bestn[mt][v]], 1);
            }
    }
}

__global__ __launch_bounds__(256) void count_low_kernel(
    const int* __restrict__ counts, float* __restrict__ out_count) {
    int k = blockIdx.x * blockDim.x + threadIdx.x;
    if (k < K_ && counts[k] == 0) atomicAdd(out_count, 1.0f);
}

__global__ __launch_bounds__(256) void epilogue_kernel(
    const float* __restrict__ cb_l, const int* __restrict__ idx,
    float* __restrict__ resid, unsigned short* __restrict__ rhi,
    unsigned short* __restrict__ rlo, float* __restrict__ zacc,
    float* __restrict__ r_out_l, float* __restrict__ e_out_l,
    float* __restrict__ zhat_out, int is_last) {
    long i = (long)blockIdx.x * blockDim.x + threadIdx.x;
    if (i >= (long)B_ * D_) return;
    int b = (int)(i >> 9);      // D_ = 512
    int d = (int)(i & (D_ - 1));
    int code = idx[b];
    float q  = cb_l[(long)code * D_ + d];
    float xr = resid[i];
    long ro = (long)b * (L_ * D_) + d;
    r_out_l[ro] = xr;
    e_out_l[ro] = q;
    float nr = xr - q;
    resid[i] = nr;
    unsigned short h = f32_to_bf16u(nr);
    rhi[i] = h;
    rlo[i] = f32_to_bf16u(nr - bf16u_to_f32(h));
    float za = zacc[i] + q;
    zacc[i] = za;
    if (is_last) zhat_out[(long)b * D_ + d] = za;
}

// ---------------- host launch ----------------

extern "C" void kernel_launch(void* const* d_in, const int* in_sizes, int n_in,
                              void* d_out, int out_size, void* d_ws,
                              size_t ws_size, hipStream_t stream) {
    (void)in_sizes; (void)n_in; (void)out_size; (void)ws_size;
    const float* x  = (const float*)d_in[0];   // [B, D]
    const float* cb = (const float*)d_in[1];   // [L, K, D]

    float* out       = (float*)d_out;
    float* codes_out = out;                                  // [B, L]
    float* r_out     = codes_out + (long)B_ * L_;            // [B, L, D]
    float* e_out     = r_out + (long)B_ * L_ * D_;           // [B, L, D]
    float* zhat_out  = e_out + (long)B_ * L_ * D_;           // [B, D]
    float* count_out = zhat_out + (long)B_ * D_;             // scalar

    char* w = (char*)d_ws;
    float* resid          = (float*)w;          w += (long)B_ * D_ * 4;
    float* zacc           = (float*)w;          w += (long)B_ * D_ * 4;
    unsigned short* rhi   = (unsigned short*)w; w += (long)B_ * D_ * 2;
    unsigned short* rlo   = (unsigned short*)w; w += (long)B_ * D_ * 2;
    unsigned short* cbhi  = (unsigned short*)w; w += (long)L_ * K_ * D_ * 2;
    unsigned short* cblo  = (unsigned short*)w; w += (long)L_ * K_ * D_ * 2;
    float* cbnorm         = (float*)w;          w += (long)L_ * K_ * 4;
    int*   idxb           = (int*)w;            w += (long)B_ * 4;
    int*   counts         = (int*)w;            w += (long)L_ * K_ * 4;

    prep_cb_kernel<<<((long)L_ * K_ * D_ + 255) / 256, 256, 0, stream>>>(
        cb, cbhi, cblo);
    prep_norm_kernel<<<(L_ * K_ + 255) / 256, 256, 0, stream>>>(
        cb, cbnorm, counts, count_out);
    prep_x_kernel<<<((long)B_ * D_ + 255) / 256, 256, 0, stream>>>(
        x, resid, rhi, rlo, zacc);

    for (int l = 0; l < L_; ++l) {
        vq_argmin_kernel<<<B_ / (8 * 16 * MT), 256, 0, stream>>>(
            (const __bf16*)rhi, (const __bf16*)rlo,
            cbhi + (long)l * K_ * D_, cblo + (long)l * K_ * D_,
            cbnorm + l * K_, idxb, counts + l * K_, codes_out + l);
        count_low_kernel<<<(K_ + 255) / 256, 256, 0, stream>>>(
            counts + l * K_, count_out);
        epilogue_kernel<<<((long)B_ * D_ + 255) / 256, 256, 0, stream>>>(
            cb + (long)l * K_ * D_, idxb, resid, rhi, rlo, zacc,
            r_out + l * D_, e_out + l * D_, zhat_out, l == L_ - 1);
    }
}
